// DIEN_56332791054405
// MI455X (gfx1250) — matrix operand order repaired and checked by reading, not compiled, not run on hardware
//
#include <hip/hip_runtime.h>
#include <hip/hip_bf16.h>
#include <math.h>

#define BB 8192
#define TT 40
#define VV 50000
#define EE 16
#define HH 48
#define GG 144   // 3*H

typedef __attribute__((ext_vector_type(16))) __bf16 v16bf;
typedef __attribute__((ext_vector_type(8)))  float  v8f;

#define WMMA(c,a,b) (c) = __builtin_amdgcn_wmma_f32_16x16x32_bf16(false,(a),false,(b),(short)0,(c),false,false)

__device__ __forceinline__ float sigf(float x){ return 1.0f/(1.0f+__expf(-x)); }

union frag_u { uint4 u[2]; v16bf v; };

// pack 4 f32 -> 4 bf16 (8 bytes)
__device__ __forceinline__ uint2 pk4(float4 f){
  union { __bf16 h[4]; uint2 u; } p;
  p.h[0] = (__bf16)f.x; p.h[1] = (__bf16)f.y; p.h[2] = (__bf16)f.z; p.h[3] = (__bf16)f.w;
  return p.u;
}

// A-matrix fragment: 16x32 bf16 tile, row-major LDS, leading dim ld (multiple of 8).
// lane holds row = lane&15; elements 0..7 <-> K = 8*half+[0..7]; 8..15 <-> K = 16+8*half+[0..7]
// => two contiguous 16-byte LDS loads per lane.
__device__ __forceinline__ v16bf frag_a(const __bf16* p, int ld, int lane){
  const __bf16* r = p + (lane & 15)*ld + ((lane >> 4) << 3);
  frag_u f;
  f.u[0] = *(const uint4*)r;
  f.u[1] = *(const uint4*)(r + 16);
  return f.v;
}

// B-matrix fragment from pre-swizzled weights: tile = 32 lanes x 16 bf16 contiguous.
__device__ __forceinline__ v16bf frag_b_sw(const __bf16* tile, int lane){
  const uint4* p = (const uint4*)(tile + lane*16);
  frag_u f;
  f.u[0] = p[0];
  f.u[1] = p[1];
  return f.v;
}

// stage KxN f32 row-major weights -> swizzled bf16 tiles [KT][NT][32 lanes][16]
// lane L of tile (kt,nt): element e <-> (k = kt*32 + 8*(L>>4) + (e&7) + 16*(e>>3), n = nt*16 + (L&15))
__device__ void stage_w_sw(__bf16* dst, const float* src, int K, int N, int KT, int NT,
                           int tid, int nth){
  int total = KT*NT*512;
  for (int i = tid; i < total; i += nth){
    int tile = i >> 9, rem = i & 511;
    int lane = rem >> 4, e = rem & 15;
    int kt = tile / NT, nt = tile - kt*NT;
    int k = kt*32 + ((lane >> 4) << 3) + (e & 7) + ((e >> 3) << 4);
    int n = nt*16 + (lane & 15);
    dst[i] = (__bf16)((k < K && n < N) ? src[k*N + n] : 0.0f);
  }
}
__device__ void stage_f(float* dst, const float* src, int N, int Npad, int tid, int nth){
  for (int i = tid; i < Npad; i += nth) dst[i] = (i < N) ? src[i] : 0.0f;
}

// ---------------- embedding gathers ----------------
__global__ void k_gather_seq(const int* __restrict__ seq, const int* __restrict__ seqn,
                             const float* __restrict__ Eseq,
                             float* __restrict__ maskf, float* __restrict__ se,
                             float* __restrict__ sen){
  int i = blockIdx.x*blockDim.x + threadIdx.x;
  if (i >= BB*TT) return;
  const int* s3 = seq  + (size_t)i*3;
  const int* n3 = seqn + (size_t)i*3;
  maskf[i] = (s3[0] != 0) ? 1.0f : 0.0f;
  float4* o  = (float4*)(se  + (size_t)i*HH);
  float4* on = (float4*)(sen + (size_t)i*HH);
#pragma unroll
  for (int j = 0; j < 3; ++j){
    const float4* p = (const float4*)(Eseq + ((size_t)j*VV + s3[j])*EE);
    const float4* q = (const float4*)(Eseq + ((size_t)j*VV + n3[j])*EE);
#pragma unroll
    for (int v = 0; v < 4; ++v){ o[j*4+v] = p[v]; on[j*4+v] = q[v]; }
  }
}

__global__ void k_gather_tgt(const int* __restrict__ tiseq, const int* __restrict__ tiside,
                             const int* __restrict__ tuser,
                             const float* __restrict__ Eseq, const float* __restrict__ Eitem,
                             const float* __restrict__ Euser,
                             float* __restrict__ tseq, float* __restrict__ tside,
                             float* __restrict__ uside){
  int b = blockIdx.x*blockDim.x + threadIdx.x;
  if (b >= BB) return;
  float4* o1 = (float4*)(tseq  + (size_t)b*HH);
  float4* o2 = (float4*)(tside + (size_t)b*HH);
  float4* o3 = (float4*)(uside + (size_t)b*80);
#pragma unroll
  for (int j = 0; j < 3; ++j){
    const float4* p = (const float4*)(Eseq  + ((size_t)j*VV + tiseq [b*3+j])*EE);
    const float4* q = (const float4*)(Eitem + ((size_t)j*VV + tiside[b*3+j])*EE);
#pragma unroll
    for (int v = 0; v < 4; ++v){ o1[j*4+v] = p[v]; o2[j*4+v] = q[v]; }
  }
#pragma unroll
  for (int j = 0; j < 5; ++j){
    const float4* p = (const float4*)(Euser + ((size_t)j*VV + tuser[b*5+j])*EE);
#pragma unroll
    for (int v = 0; v < 4; ++v) o3[j*4+v] = p[v];
  }
}

// ---------------- GRU / AUGRU (wave = 16 batch rows, h carried in accumulators) ----------------
template<bool AUGRU>
__global__ __launch_bounds__(64)
void k_rnn(const float* __restrict__ xin, const float* __restrict__ Wx,
           const float* __restrict__ Wh, const float* __restrict__ bin,
           const float* __restrict__ brec, const float* __restrict__ maskf,
           const float* __restrict__ att, float* __restrict__ yout){
  __shared__ __align__(16) __bf16 sWx[2*9*512];   // KT=2, NT=9 swizzled
  __shared__ __align__(16) __bf16 sWh[2*9*512];
  __shared__ float  sb[2*GG];
  __shared__ __align__(16) __bf16 sX[2][16*64];
  __shared__ __align__(16) __bf16 sH[2][16*64];
  int tid = threadIdx.x, nth = blockDim.x;
  int wid = tid >> 5, lane = tid & 31;
  stage_w_sw(sWx, Wx, HH, GG, 2, 9, tid, nth);
  stage_w_sw(sWh, Wh, HH, GG, 2, 9, tid, nth);
  for (int i = tid; i < GG; i += nth){ sb[i] = bin[i]; sb[GG+i] = AUGRU ? 0.0f : brec[i]; }
  for (int i = tid; i < 2*16*64; i += nth){
    ((__bf16*)sH)[i] = (__bf16)0.0f; ((__bf16*)sX)[i] = (__bf16)0.0f;
  }
  __syncthreads();
  int b0 = blockIdx.x*32 + wid*16;
  int n = lane & 15, half = lane >> 4;
  v8f hacc[3];
#pragma unroll
  for (int i = 0; i < 3; ++i)
#pragma unroll
    for (int j = 0; j < 8; ++j) hacc[i][j] = 0.0f;

  for (int t = 0; t < TT; ++t){
    // stage x tile: float4 loads -> packed bf16 b64 stores (12 float4 per row)
    for (int idx = lane; idx < 16*12; idx += 32){
      int r = idx/12, c4 = (idx - r*12)*4;
      float4 f = *(const float4*)&xin[((size_t)(b0+r)*TT + t)*HH + c4];
      *(uint2*)&sX[wid][r*64 + c4] = pk4(f);
    }
    __syncthreads();
    v8f zr[6], xh[3], hh[3];
#pragma unroll
    for (int i = 0; i < 6; ++i){ float bv = sb[i*16+n] + sb[GG + i*16+n];
#pragma unroll
      for (int j = 0; j < 8; ++j) zr[i][j] = bv; }
#pragma unroll
    for (int i = 0; i < 3; ++i){ float b1v = sb[96+i*16+n], b2v = sb[GG+96+i*16+n];
#pragma unroll
      for (int j = 0; j < 8; ++j){ xh[i][j] = b1v; hh[i][j] = b2v; } }
#pragma unroll
    for (int kc = 0; kc < 2; ++kc){
      v16bf ax = frag_a(sX[wid] + kc*32, 64, lane);
      v16bf ah = frag_a(sH[wid] + kc*32, 64, lane);
#pragma unroll
      for (int nt = 0; nt < 9; ++nt){
        v16bf bx = frag_b_sw(sWx + ((kc*9 + nt) << 9), lane);
        v16bf bh = frag_b_sw(sWh + ((kc*9 + nt) << 9), lane);
        if (nt < 6){ WMMA(zr[nt], ax, bx); WMMA(zr[nt], ah, bh); }
        else       { WMMA(xh[nt-6], ax, bx); WMMA(hh[nt-6], ah, bh); }
      }
    }
    float mrow[8], arow[8];
#pragma unroll
    for (int j = 0; j < 8; ++j){
      int row = half*8 + j;
      mrow[j] = maskf[(size_t)(b0+row)*TT + t];
      arow[j] = AUGRU ? att[(size_t)(b0+row)*TT + t] : 0.0f;
    }
#pragma unroll
    for (int i = 0; i < 3; ++i){
#pragma unroll
      for (int j = 0; j < 8; ++j){
        float zg = sigf(zr[i][j]);
        float rg = sigf(zr[3+i][j]);
        float cand = tanhf(xh[i][j] + rg*hh[i][j]);
        float h_old = hacc[i][j];
        float hn;
        if (AUGRU){ float z2 = arow[j]*zg; hn = (1.0f - z2)*h_old + z2*cand; }
        else      { hn = zg*h_old + (1.0f - zg)*cand; }
        float hv = mrow[j]*hn + (1.0f - mrow[j])*h_old;
        hacc[i][j] = hv;
        int row = half*8 + j;
        sH[wid][row*64 + i*16 + n] = (__bf16)hv;
        if (!AUGRU) yout[((size_t)(b0+row)*TT + t)*HH + i*16 + n] = hv;
      }
    }
    __syncthreads();
  }
  if (AUGRU){
#pragma unroll
    for (int i = 0; i < 3; ++i)
#pragma unroll
      for (int j = 0; j < 8; ++j){
        int row = half*8 + j;
        yout[(size_t)(b0+row)*HH + i*16 + n] = hacc[i][j];
      }
  }
}

// ---------------- auxiliary MLP + loss partials ----------------
__global__ __launch_bounds__(64)
void k_aux(const float* __restrict__ gru, const float* __restrict__ se,
           const float* __restrict__ sen, const float* __restrict__ maskf,
           const float* __restrict__ W1, const float* __restrict__ b1,
           const float* __restrict__ W2, const float* __restrict__ b2,
           const float* __restrict__ W3, const float* __restrict__ b3,
           float* __restrict__ part){
  __shared__ __align__(16) __bf16 sW1[3*7*512];   // K 96, N 100->112
  __shared__ __align__(16) __bf16 sW2[4*4*512];   // K 100->128, N 50->64
  __shared__ float  sb1[112], sb2[64], sW3[50], sb3;
  __shared__ __align__(16) __bf16 sA[2][16*96];
  __shared__ __align__(16) __bf16 sAct[2][16*128];
  __shared__ float  sRed[2][16*64];
  __shared__ float  sLoss[2][16];
  int tid = threadIdx.x, nth = blockDim.x, wid = tid >> 5, lane = tid & 31;
  stage_w_sw(sW1, W1, 96, 100, 3, 7, tid, nth);
  stage_w_sw(sW2, W2, 100, 50, 4, 4, tid, nth);
  stage_f(sb1, b1, 100, 112, tid, nth);
  stage_f(sb2, b2, 50, 64, tid, nth);
  stage_f(sW3, W3, 50, 50, tid, nth);
  if (tid == 0) sb3 = b3[0];
  for (int i = tid; i < 2*16*128; i += nth) ((__bf16*)sAct)[i] = (__bf16)0.0f;
  __syncthreads();
  int n = lane & 15, half = lane >> 4;
  int r0 = (blockIdx.x*2 + wid)*16;
  float wloss = 0.0f;
#pragma unroll
  for (int pass = 0; pass < 2; ++pass){
    const float* xsrc = pass ? sen : se;
    for (int idx = lane; idx < 16*24; idx += 32){
      int r = idx/24, q = idx - r*24;
      int rr = r0 + r, b = rr/(TT-1), tm = rr - b*(TT-1);
      const float* src = (q < 12) ? &gru [((size_t)b*TT + tm  )*HH + q*4]
                                  : &xsrc[((size_t)b*TT + tm+1)*HH + (q-12)*4];
      *(uint2*)&sA[wid][r*96 + q*4] = pk4(*(const float4*)src);
    }
    __syncthreads();
    v8f acc1[7];
#pragma unroll
    for (int i = 0; i < 7; ++i){ float bv = sb1[i*16+n];
#pragma unroll
      for (int j = 0; j < 8; ++j) acc1[i][j] = bv; }
#pragma unroll
    for (int kc = 0; kc < 3; ++kc){
      v16bf a = frag_a(sA[wid] + kc*32, 96, lane);
#pragma unroll
      for (int nt = 0; nt < 7; ++nt){
        v16bf bw = frag_b_sw(sW1 + ((kc*7 + nt) << 9), lane);
        WMMA(acc1[nt], a, bw);
      }
    }
#pragma unroll
    for (int i = 0; i < 7; ++i)
#pragma unroll
      for (int j = 0; j < 8; ++j)
        sAct[wid][(half*8+j)*128 + i*16 + n] = (__bf16)sigf(acc1[i][j]);
    __syncthreads();
    v8f acc2[4];
#pragma unroll
    for (int i = 0; i < 4; ++i){ float bv = sb2[i*16+n];
#pragma unroll
      for (int j = 0; j < 8; ++j) acc2[i][j] = bv; }
#pragma unroll
    for (int kc = 0; kc < 4; ++kc){
      v16bf a = frag_a(sAct[wid] + kc*32, 128, lane);
#pragma unroll
      for (int nt = 0; nt < 4; ++nt){
        v16bf bw = frag_b_sw(sW2 + ((kc*4 + nt) << 9), lane);
        WMMA(acc2[nt], a, bw);
      }
    }
#pragma unroll
    for (int i = 0; i < 4; ++i)
#pragma unroll
      for (int j = 0; j < 8; ++j)
        sRed[wid][(half*8+j)*64 + i*16 + n] = sigf(acc2[i][j]);
    __syncthreads();
    if (lane < 16){
      int rr = r0 + lane, b = rr/(TT-1), tm = rr - b*(TT-1);
      float m = maskf[(size_t)b*TT + tm + 1];
      float s = sb3;
      for (int c = 0; c < 50; ++c) s += sRed[wid][lane*64 + c]*sW3[c];
      float p = sigf(s);
      sLoss[wid][lane] = pass ? -__logf(1.0f - p + 1e-8f)*m : -__logf(p + 1e-8f)*m;
    }
    __syncthreads();
    if (lane == 0){ float s = 0.0f; for (int k = 0; k < 16; ++k) s += sLoss[wid][k]; wloss += s; }
    __syncthreads();
  }
  if (lane == 0) part[blockIdx.x*2 + wid] = wloss;
}

// ---------------- attention score MLP ----------------
__global__ __launch_bounds__(64)
void k_attsc(const float* __restrict__ gru, const float* __restrict__ tseq,
             const float* __restrict__ W1, const float* __restrict__ b1,
             const float* __restrict__ W2, const float* __restrict__ b2,
             const float* __restrict__ Wo, const float* __restrict__ bo,
             float* __restrict__ scores){
  __shared__ __align__(16) __bf16 sW1[6*5*512];   // K 192, N 80
  __shared__ __align__(16) __bf16 sW2[3*3*512];   // K 80->96, N 40->48
  __shared__ float sb1[80], sb2[48], sWo[40], sbo;
  __shared__ __align__(16) __bf16 sA[2][16*192];
  __shared__ __align__(16) __bf16 sAct[2][16*96];
  int tid = threadIdx.x, nth = blockDim.x, wid = tid >> 5, lane = tid & 31;
  stage_w_sw(sW1, W1, 192, 80, 6, 5, tid, nth);
  stage_w_sw(sW2, W2, 80, 40, 3, 3, tid, nth);
  stage_f(sb1, b1, 80, 80, tid, nth);
  stage_f(sb2, b2, 40, 48, tid, nth);
  stage_f(sWo, Wo, 40, 40, tid, nth);
  if (tid == 0) sbo = bo[0];
  for (int i = tid; i < 2*16*96; i += nth) ((__bf16*)sAct)[i] = (__bf16)0.0f;
  __syncthreads();
  int n = lane & 15, half = lane >> 4;
  int r0 = (blockIdx.x*2 + wid)*16;
  // build info = [q, k, q-k, q*k] per row: load q4/k4 once, emit 4 packed stores
  for (int idx = lane; idx < 16*12; idx += 32){
    int r = idx/12, cc4 = (idx - r*12)*4;
    int rr = r0 + r, b = rr/TT, t = rr - b*TT;
    float4 q = *(const float4*)&tseq[(size_t)b*HH + cc4];
    float4 k = *(const float4*)&gru[((size_t)b*TT + t)*HH + cc4];
    float4 d; d.x = q.x-k.x; d.y = q.y-k.y; d.z = q.z-k.z; d.w = q.w-k.w;
    float4 m; m.x = q.x*k.x; m.y = q.y*k.y; m.z = q.z*k.z; m.w = q.w*k.w;
    __bf16* row = &sA[wid][r*192];
    *(uint2*)&row[cc4]       = pk4(q);
    *(uint2*)&row[48 + cc4]  = pk4(k);
    *(uint2*)&row[96 + cc4]  = pk4(d);
    *(uint2*)&row[144 + cc4] = pk4(m);
  }
  __syncthreads();
  v8f acc1[5];
#pragma unroll
  for (int i = 0; i < 5; ++i){ float bv = sb1[i*16+n];
#pragma unroll
    for (int j = 0; j < 8; ++j) acc1[i][j] = bv; }
#pragma unroll
  for (int kc = 0; kc < 6; ++kc){
    v16bf a = frag_a(sA[wid] + kc*32, 192, lane);
#pragma unroll
    for (int nt = 0; nt < 5; ++nt){
      v16bf bw = frag_b_sw(sW1 + ((kc*5 + nt) << 9), lane);
      WMMA(acc1[nt], a, bw);
    }
  }
#pragma unroll
  for (int i = 0; i < 5; ++i)
#pragma unroll
    for (int j = 0; j < 8; ++j)
      sAct[wid][(half*8+j)*96 + i*16 + n] = (__bf16)sigf(acc1[i][j]);
  __syncthreads();
  v8f acc2[3];
#pragma unroll
  for (int i = 0; i < 3; ++i){ float bv = sb2[i*16+n];
#pragma unroll
    for (int j = 0; j < 8; ++j) acc2[i][j] = bv; }
#pragma unroll
  for (int kc = 0; kc < 3; ++kc){
    v16bf a = frag_a(sAct[wid] + kc*32, 96, lane);
#pragma unroll
    for (int nt = 0; nt < 3; ++nt){
      v16bf bw = frag_b_sw(sW2 + ((kc*3 + nt) << 9), lane);
      WMMA(acc2[nt], a, bw);
    }
  }
  float* sRed = (float*)&sA[wid][0];   // 16x48 f32 fits in 16x192 bf16 region
#pragma unroll
  for (int i = 0; i < 3; ++i)
#pragma unroll
    for (int j = 0; j < 8; ++j)
      sRed[(half*8+j)*48 + i*16 + n] = sigf(acc2[i][j]);
  __syncthreads();
  if (lane < 16){
    float s = sbo;
    for (int c = 0; c < 40; ++c) s += sRed[lane*48 + c]*sWo[c];
    scores[r0 + lane] = s;
  }
}

// ---------------- masked softmax over T ----------------
__global__ void k_softmax(const float* __restrict__ scores, const float* __restrict__ maskf,
                          float* __restrict__ att){
  __shared__ float sv[TT];
  __shared__ float red[2];
  int b = blockIdx.x, tid = threadIdx.x;
  if (tid < TT){
    float m = maskf[(size_t)b*TT + tid];
    sv[tid] = (m > 0.0f) ? scores[(size_t)b*TT + tid] : -4294967296.0f;
  }
  __syncthreads();
  if (tid == 0){ float mx = sv[0]; for (int i = 1; i < TT; ++i) mx = fmaxf(mx, sv[i]); red[0] = mx; }
  __syncthreads();
  if (tid < TT) sv[tid] = __expf(sv[tid] - red[0]);
  __syncthreads();
  if (tid == 0){ float s = 0.0f; for (int i = 0; i < TT; ++i) s += sv[i]; red[1] = s; }
  __syncthreads();
  if (tid < TT) att[(size_t)b*TT + tid] = sv[tid]/red[1];
}

// ---------------- final BN + FFN + output ----------------
__global__ __launch_bounds__(32)
void k_final(const float* __restrict__ hfin, const float* __restrict__ tseq,
             const float* __restrict__ tside, const float* __restrict__ uside,
             const float* __restrict__ bng, const float* __restrict__ bnb,
             const float* __restrict__ bnm, const float* __restrict__ bnv,
             const float* __restrict__ W1, const float* __restrict__ b1,
             const float* __restrict__ a1, const float* __restrict__ W2,
             const float* __restrict__ b2, const float* __restrict__ a2,
             const float* __restrict__ Wo, const float* __restrict__ bo,
             float* __restrict__ outp){
  __shared__ __align__(16) __bf16 sW1[7*5*512];   // K 224, N 80
  __shared__ __align__(16) __bf16 sW2[3*3*512];   // K 80->96, N 40->48
  __shared__ float sb1[80], sa1[80], sb2[48], sa2[48], sWo[80], sbo[2];
  __shared__ __align__(16) float sScale[224], sShift[224];
  __shared__ __align__(16) __bf16 sA[16*224];
  __shared__ __align__(16) __bf16 sAct[16*96];
  int tid = threadIdx.x, nth = blockDim.x, lane = tid;
  stage_w_sw(sW1, W1, 224, 80, 7, 5, tid, nth);
  stage_w_sw(sW2, W2, 80, 40, 3, 3, tid, nth);
  stage_f(sb1, b1, 80, 80, tid, nth); stage_f(sa1, a1, 80, 80, tid, nth);
  stage_f(sb2, b2, 40, 48, tid, nth); stage_f(sa2, a2, 40, 48, tid, nth);
  stage_f(sWo, Wo, 80, 80, tid, nth);
  if (tid < 2) sbo[tid] = bo[tid];
  for (int i = tid; i < 224; i += nth){
    float s = bng[i]*rsqrtf(bnv[i] + 0.001f);
    sScale[i] = s;
    sShift[i] = bnb[i] - bnm[i]*s;
  }
  for (int i = tid; i < 16*96; i += nth) sAct[i] = (__bf16)0.0f;
  __syncthreads();
  int n = lane & 15, half = lane >> 4;
  int b0 = blockIdx.x*16;
  for (int idx = lane; idx < 16*56; idx += 32){
    int r = idx/56, c4 = (idx - r*56)*4;
    int b = b0 + r;
    const float* src;
    if      (c4 < 48)  src = &hfin [(size_t)b*HH + c4];
    else if (c4 < 96)  src = &tseq [(size_t)b*HH + c4 - 48];
    else if (c4 < 144) src = &tside[(size_t)b*HH + c4 - 96];
    else               src = &uside[(size_t)b*80 + c4 - 144];
    float4 x  = *(const float4*)src;
    float4 sc = *(const float4*)&sScale[c4];
    float4 sh = *(const float4*)&sShift[c4];
    float4 y;
    y.x = x.x*sc.x + sh.x; y.y = x.y*sc.y + sh.y;
    y.z = x.z*sc.z + sh.z; y.w = x.w*sc.w + sh.w;
    *(uint2*)&sA[r*224 + c4] = pk4(y);
  }
  __syncthreads();
  v8f acc1[5];
#pragma unroll
  for (int i = 0; i < 5; ++i){ float bv = sb1[i*16+n];
#pragma unroll
    for (int j = 0; j < 8; ++j) acc1[i][j] = bv; }
#pragma unroll
  for (int kc = 0; kc < 7; ++kc){
    v16bf a = frag_a(sA + kc*32, 224, lane);
#pragma unroll
    for (int nt = 0; nt < 5; ++nt){
      v16bf bw = frag_b_sw(sW1 + ((kc*5 + nt) << 9), lane);
      WMMA(acc1[nt], a, bw);
    }
  }
#pragma unroll
  for (int i = 0; i < 5; ++i){
    float al = sa1[i*16+n];
#pragma unroll
    for (int j = 0; j < 8; ++j){
      float x = acc1[i][j];
      sAct[(half*8+j)*96 + i*16 + n] = (__bf16)((x >= 0.0f) ? x : al*x);
    }
  }
  __syncthreads();
  v8f acc2[3];
#pragma unroll
  for (int i = 0; i < 3; ++i){ float bv = sb2[i*16+n];
#pragma unroll
    for (int j = 0; j < 8; ++j) acc2[i][j] = bv; }
#pragma unroll
  for (int kc = 0; kc < 3; ++kc){
    v16bf a = frag_a(sAct + kc*32, 96, lane);
#pragma unroll
    for (int nt = 0; nt < 3; ++nt){
      v16bf bw = frag_b_sw(sW2 + ((kc*3 + nt) << 9), lane);
      WMMA(acc2[nt], a, bw);
    }
  }
  float* sRed = (float*)&sA[0];
#pragma unroll
  for (int i = 0; i < 3; ++i){
    float al = sa2[i*16+n];
#pragma unroll
    for (int j = 0; j < 8; ++j){
      float x = acc2[i][j];
      sRed[(half*8+j)*48 + i*16 + n] = (x >= 0.0f) ? x : al*x;
    }
  }
  __syncthreads();
  {
    int r = lane >> 1, o = lane & 1;
    float z = sbo[o];
    for (int c = 0; c < 40; ++c) z += sRed[r*48 + c]*sWo[c*2 + o];
    int b = b0 + r;
    outp[(size_t)b*2 + o] = sigf(z);             // outputs
    outp[(size_t)BB*2 + (size_t)b*2 + o] = z;    // logits
  }
}

// ---------------- deterministic aux-loss reduction ----------------
__global__ void k_auxfin(const float* __restrict__ part, int nparts, float* __restrict__ outp){
  __shared__ float sred[256];
  int tid = threadIdx.x;
  float s = 0.0f;
  for (int i = tid; i < nparts; i += 256) s += part[i];
  sred[tid] = s; __syncthreads();
  for (int st = 128; st > 0; st >>= 1){
    if (tid < st) sred[tid] += sred[tid + st];
    __syncthreads();
  }
  if (tid == 0) outp[(size_t)BB*4] = sred[0] / (float)((size_t)BB*(TT-1));
}

extern "C" void kernel_launch(void* const* d_in, const int* in_sizes, int n_in,
                              void* d_out, int out_size, void* d_ws, size_t ws_size,
                              hipStream_t stream) {
  (void)in_sizes; (void)n_in; (void)out_size; (void)ws_size;
  const int*   tuser  = (const int*)d_in[1];
  const int*   seqi   = (const int*)d_in[2];
  const int*   seqni  = (const int*)d_in[3];
  const int*   tiseq  = (const int*)d_in[4];
  const int*   tiside = (const int*)d_in[5];
  const float* Eseq   = (const float*)d_in[6];
  const float* Euser  = (const float*)d_in[7];
  const float* Eitem  = (const float*)d_in[8];
  const float* gWx    = (const float*)d_in[9];
  const float* gWh    = (const float*)d_in[10];
  const float* gbin   = (const float*)d_in[11];
  const float* gbrec  = (const float*)d_in[12];
  const float* auWx   = (const float*)d_in[13];
  const float* auWh   = (const float*)d_in[14];
  const float* aub    = (const float*)d_in[15];
  const float* aW1    = (const float*)d_in[16];
  const float* ab1    = (const float*)d_in[17];
  const float* aW2    = (const float*)d_in[18];
  const float* ab2    = (const float*)d_in[19];
  const float* aWo    = (const float*)d_in[20];
  const float* abo    = (const float*)d_in[21];
  const float* xW1    = (const float*)d_in[22];
  const float* xb1    = (const float*)d_in[23];
  const float* xW2    = (const float*)d_in[24];
  const float* xb2    = (const float*)d_in[25];
  const float* xW3    = (const float*)d_in[26];
  const float* xb3    = (const float*)d_in[27];
  const float* bng    = (const float*)d_in[28];
  const float* bnb    = (const float*)d_in[29];
  const float* bnm    = (const float*)d_in[30];
  const float* bnv    = (const float*)d_in[31];
  const float* fW1    = (const float*)d_in[32];
  const float* fb1    = (const float*)d_in[33];
  const float* al1    = (const float*)d_in[34];
  const float* fW2    = (const float*)d_in[35];
  const float* fb2    = (const float*)d_in[36];
  const float* al2    = (const float*)d_in[37];
  const float* oW     = (const float*)d_in[38];
  const float* ob     = (const float*)d_in[39];

  float* w = (float*)d_ws;
  size_t off = 0;
  float* maskf  = w + off; off += (size_t)BB*TT;
  float* se     = w + off; off += (size_t)BB*TT*HH;
  float* sen    = w + off; off += (size_t)BB*TT*HH;
  float* gru    = w + off; off += (size_t)BB*TT*HH;
  float* tseq   = w + off; off += (size_t)BB*HH;
  float* tside  = w + off; off += (size_t)BB*HH;
  float* uside  = w + off; off += (size_t)BB*80;
  float* scores = w + off; off += (size_t)BB*TT;
  float* attw   = w + off; off += (size_t)BB*TT;
  float* hfin   = w + off; off += (size_t)BB*HH;
  float* auxp   = w + off; off += (size_t)BB*(TT-1)/16;   // 19968 per-wave partials

  k_gather_seq<<<(BB*TT + 255)/256, 256, 0, stream>>>(seqi, seqni, Eseq, maskf, se, sen);
  k_gather_tgt<<<(BB + 255)/256, 256, 0, stream>>>(tiseq, tiside, tuser, Eseq, Eitem, Euser,
                                                   tseq, tside, uside);
  k_rnn<false><<<BB/32, 64, 0, stream>>>(se, gWx, gWh, gbin, gbrec, maskf, nullptr, gru);
  k_aux<<<BB*(TT-1)/32, 64, 0, stream>>>(gru, se, sen, maskf, xW1, xb1, xW2, xb2, xW3, xb3, auxp);
  k_attsc<<<BB*TT/32, 64, 0, stream>>>(gru, tseq, aW1, ab1, aW2, ab2, aWo, abo, scores);
  k_softmax<<<BB, 64, 0, stream>>>(scores, maskf, attw);
  k_rnn<true><<<BB/32, 64, 0, stream>>>(gru, auWx, auWh, aub, nullptr, maskf, attw, hfin);
  k_final<<<BB/16, 32, 0, stream>>>(hfin, tseq, tside, uside, bng, bnb, bnm, bnv,
                                    fW1, fb1, al1, fW2, fb2, al2, oW, ob, (float*)d_out);
  k_auxfin<<<1, 256, 0, stream>>>(auxp, BB*(TT-1)/16, (float*)d_out);
}
